// Net_79680233276088
// MI455X (gfx1250) — compile-verified, matrix operand order checked by reading, not run on hardware
//
#include <hip/hip_runtime.h>

// CDNA5 / gfx1250 GCN forward pass.
// Dense transforms: v_wmma_f32_16x16x32_bf16, K/DOUT compile-time, B pre-packed
// into the WMMA B-operand lane layout (bf16), A loaded as contiguous float4 and
// converted inline. Edge aggregation: float4 gathers + f32 global atomics
// (bandwidth-bound part, ~3.3 GB total traffic -> ~140us floor at 23.3 TB/s).

typedef __attribute__((ext_vector_type(16))) __bf16 v16bf;
typedef __attribute__((ext_vector_type(8)))  float  v8f;

#define WAVES_PER_BLOCK 4
#define N_GRAPHS 64
#define NUM_CLASSES 10
#define D_FINAL 256

// ---------------- utility kernels ----------------

__global__ void k_fill(float* __restrict__ p, float v, int n) {
    int i = blockIdx.x * blockDim.x + threadIdx.x;
    if (i < n) p[i] = v;
}

__global__ void k_deg_acc(const int* __restrict__ dst, float* __restrict__ deg, int E) {
    int e = blockIdx.x * blockDim.x + threadIdx.x;
    if (e < E) atomicAdd(&deg[dst[e]], 1.0f);
}

__global__ void k_rsqrt_inplace(float* __restrict__ p, int n) {
    int i = blockIdx.x * blockDim.x + threadIdx.x;
    if (i < n) p[i] = rsqrtf(p[i]);
}

// Zero-pad x[n x 5] -> xpad[n x 32] (f32) so layer 1 shares the K=32 WMMA path.
__global__ void k_pad_x(const float* __restrict__ x, float* __restrict__ xpad, int n) {
    long long idx = (long long)blockIdx.x * blockDim.x + threadIdx.x;
    long long total = (long long)n * 32;
    if (idx >= total) return;
    int i = (int)(idx >> 5);
    int c = (int)(idx & 31);
    xpad[idx] = (c < 5) ? x[(size_t)i * 5 + c] : 0.0f;
}

// ---------------- W pre-pack into WMMA B-operand layout ----------------
// B (32x16 bf16 per K-step, ISA 7.12.2): lane&15 = N column, lanes>=16 take the
// K+8 half; vector elems 0..7 -> K = kbase..kbase+7, elems 8..15 -> kbase+16..+23.
// Packed layout: Wp[((ks*tilesN + tn)*32 + lane)*16 + i], so each lane loads its
// 16 bf16 B values as one aligned 32-byte vector.
template <int KPAD, int DOUT>
__global__ void k_pack_w(const float* __restrict__ W, __bf16* __restrict__ Wp, int Kreal) {
    constexpr int tilesN = DOUT / 16;
    constexpr int total  = (KPAD / 32) * tilesN * 32 * 16;
    int idx = blockIdx.x * blockDim.x + threadIdx.x;
    if (idx >= total) return;
    int i    = idx & 15;
    int lane = (idx >> 4) & 31;
    int tn   = (idx >> 9) % tilesN;
    int ks   = idx / (512 * tilesN);
    int kHalf = (lane >> 4) * 8;
    int col   = tn * 16 + (lane & 15);
    int k     = ks * 32 + kHalf + (i < 8 ? i : i + 8);
    Wp[idx] = (k < Kreal) ? (__bf16)W[(size_t)k * DOUT + col] : (__bf16)0.0f;
}

// ---------------- WMMA GEMM: T[n x DOUT] = H[n x K] * W[K x DOUT] ----------------
// One wave per 16x16 tile of T; K-loop fully unrolled. Requires n % 16 == 0
// (n = 50000 = 3125 * 16) and K % 32 == 0.
template <int K, int DOUT>
__global__ void k_gemm_wmma_bf16(const float* __restrict__ H, const __bf16* __restrict__ Wp,
                                 float* __restrict__ T, int n) {
    constexpr int tilesN = DOUT / 16;
    const int lane = threadIdx.x & 31;
    const int wave = blockIdx.x * WAVES_PER_BLOCK + (threadIdx.x >> 5);
    const int tilesM = n >> 4;
    if (wave >= tilesM * tilesN) return;   // wave-uniform: EXEC all-ones for WMMA

    const int tm    = wave / tilesN;
    const int tn    = wave - tm * tilesN;
    const int l15   = lane & 15;
    const int kHalf = (lane >> 4) * 8;     // 0 for lanes 0-15, 8 for lanes 16-31
    const float* hrow = H + (size_t)(tm * 16 + l15) * K;
    const __bf16* wbase = Wp + ((size_t)tn * 32 + lane) * 16;

    v8f acc = {};
#pragma unroll
    for (int ks = 0; ks < K / 32; ++ks) {
        const int kb = ks * 32 + kHalf;
        // A: two contiguous 8-float runs (K=kb..kb+7 and kb+16..kb+23), b128 loads.
        const float4 a0 = *(const float4*)(hrow + kb);
        const float4 a1 = *(const float4*)(hrow + kb + 4);
        const float4 a2 = *(const float4*)(hrow + kb + 16);
        const float4 a3 = *(const float4*)(hrow + kb + 20);
        v16bf a;
        a[0]  = (__bf16)a0.x; a[1]  = (__bf16)a0.y; a[2]  = (__bf16)a0.z; a[3]  = (__bf16)a0.w;
        a[4]  = (__bf16)a1.x; a[5]  = (__bf16)a1.y; a[6]  = (__bf16)a1.z; a[7]  = (__bf16)a1.w;
        a[8]  = (__bf16)a2.x; a[9]  = (__bf16)a2.y; a[10] = (__bf16)a2.z; a[11] = (__bf16)a2.w;
        a[12] = (__bf16)a3.x; a[13] = (__bf16)a3.y; a[14] = (__bf16)a3.z; a[15] = (__bf16)a3.w;
        // B: one aligned 32-byte vector per lane from the packed buffer.
        const v16bf b = *(const v16bf*)(wbase + (size_t)ks * tilesN * 512);
        acc = __builtin_amdgcn_wmma_f32_16x16x32_bf16(
            /*neg_a=*/false, a, /*neg_b=*/false, b,
            /*c_mod=*/(short)0, acc, /*reuse_a=*/false, /*reuse_b=*/false);
    }

    // C/D layout: VGPR i -> row = 16*tm + 8*(lane>=16) + i, col = 16*tn + (lane&15).
    const int col   = tn * 16 + l15;
    const int rbase = tm * 16 + (lane >> 4) * 8;
    float* out = T + (size_t)rbase * DOUT + col;
#pragma unroll
    for (int i = 0; i < 8; ++i) out[(size_t)i * DOUT] = acc[i];
}

// ---------------- aggregation kernels ----------------

// agg[i] = T[i] * dinv[i]^2   (self-loop contribution; also initializes agg)
__global__ void k_self_init(const float* __restrict__ t, const float* __restrict__ dinv,
                            float* __restrict__ agg, int n, int d) {
    const int chunks = d >> 2;
    long long idx = (long long)blockIdx.x * blockDim.x + threadIdx.x;
    long long total = (long long)n * chunks;
    if (idx >= total) return;
    int i  = (int)(idx / chunks);
    int c4 = (int)(idx - (long long)i * chunks) << 2;
    float w = dinv[i]; w *= w;
    float4 v = *(const float4*)(t + (size_t)i * d + c4);
    float4 o = make_float4(v.x * w, v.y * w, v.z * w, v.w * w);
    *(float4*)(agg + (size_t)i * d + c4) = o;
}

// agg[dst[e]] += T[src[e]] * dinv[src]*dinv[dst]
__global__ void k_scatter(const int* __restrict__ src, const int* __restrict__ dst,
                          const float* __restrict__ dinv, const float* __restrict__ t,
                          float* __restrict__ agg, int E, int d) {
    const int chunks = d >> 2;
    long long idx = (long long)blockIdx.x * blockDim.x + threadIdx.x;
    long long total = (long long)E * chunks;
    if (idx >= total) return;
    int e  = (int)(idx / chunks);
    int c4 = (int)(idx - (long long)e * chunks) << 2;
    int s  = src[e];
    int dn = dst[e];
    float nrm = dinv[s] * dinv[dn];
    float4 v = *(const float4*)(t + (size_t)s * d + c4);
    float* o = agg + (size_t)dn * d + c4;
    atomicAdd(o + 0, v.x * nrm);
    atomicAdd(o + 1, v.y * nrm);
    atomicAdd(o + 2, v.z * nrm);
    atomicAdd(o + 3, v.w * nrm);
}

// h = relu(agg + b), in place
__global__ void k_bias_relu(float* __restrict__ h, const float* __restrict__ b, int n, int d) {
    const int chunks = d >> 2;
    long long idx = (long long)blockIdx.x * blockDim.x + threadIdx.x;
    long long total = (long long)n * chunks;
    if (idx >= total) return;
    int i  = (int)(idx / chunks);
    int c4 = (int)(idx - (long long)i * chunks) << 2;
    float4 v = *(float4*)(h + (size_t)i * d + c4);
    v.x = fmaxf(v.x + b[c4 + 0], 0.0f);
    v.y = fmaxf(v.y + b[c4 + 1], 0.0f);
    v.z = fmaxf(v.z + b[c4 + 2], 0.0f);
    v.w = fmaxf(v.w + b[c4 + 3], 0.0f);
    *(float4*)(h + (size_t)i * d + c4) = v;
}

// ---------------- pooling + head ----------------

__global__ void k_count(const int* __restrict__ batch, float* __restrict__ counts, int n) {
    int i = blockIdx.x * blockDim.x + threadIdx.x;
    if (i < n) atomicAdd(&counts[batch[i]], 1.0f);
}

__global__ void k_pool(const int* __restrict__ batch, const float* __restrict__ h,
                       float* __restrict__ pooled, int n) {
    const int chunks = D_FINAL >> 2;  // 64
    long long idx = (long long)blockIdx.x * blockDim.x + threadIdx.x;
    long long total = (long long)n * chunks;
    if (idx >= total) return;
    int i  = (int)(idx / chunks);
    int c4 = (int)(idx - (long long)i * chunks) << 2;
    int g  = batch[i];
    float4 v = *(const float4*)(h + (size_t)i * D_FINAL + c4);
    float* o = pooled + (size_t)g * D_FINAL + c4;
    atomicAdd(o + 0, v.x);
    atomicAdd(o + 1, v.y);
    atomicAdd(o + 2, v.z);
    atomicAdd(o + 3, v.w);
}

__global__ void k_head(const float* __restrict__ pooled, const float* __restrict__ counts,
                       const float* __restrict__ fcW, const float* __restrict__ fcb,
                       float* __restrict__ out) {
    int g = blockIdx.x * blockDim.x + threadIdx.x;
    if (g >= N_GRAPHS) return;
    float inv = 1.0f / fmaxf(counts[g], 1.0f);
    float logits[NUM_CLASSES];
#pragma unroll
    for (int c = 0; c < NUM_CLASSES; ++c) logits[c] = fcb[c];
    for (int k = 0; k < D_FINAL; ++k) {
        float p = pooled[(size_t)g * D_FINAL + k] * inv;
#pragma unroll
        for (int c = 0; c < NUM_CLASSES; ++c)
            logits[c] = fmaf(p, fcW[(size_t)k * NUM_CLASSES + c], logits[c]);
    }
    float m = logits[0];
#pragma unroll
    for (int c = 1; c < NUM_CLASSES; ++c) m = fmaxf(m, logits[c]);
    float s = 0.0f;
#pragma unroll
    for (int c = 0; c < NUM_CLASSES; ++c) s += expf(logits[c] - m);
    float lse = logf(s);
#pragma unroll
    for (int c = 0; c < NUM_CLASSES; ++c)
        out[(size_t)g * NUM_CLASSES + c] = logits[c] - m - lse;
}

// ---------------- launch ----------------

extern "C" void kernel_launch(void* const* d_in, const int* in_sizes, int n_in,
                              void* d_out, int out_size, void* d_ws, size_t ws_size,
                              hipStream_t stream) {
    const float* x     = (const float*)d_in[0];
    const int*   edge  = (const int*)d_in[1];   // [2, E] flat: src row then dst row
    const int*   batch = (const int*)d_in[2];
    const float* Wl[4] = {(const float*)d_in[3], (const float*)d_in[5],
                          (const float*)d_in[7], (const float*)d_in[9]};
    const float* bl[4] = {(const float*)d_in[4], (const float*)d_in[6],
                          (const float*)d_in[8], (const float*)d_in[10]};
    const float* fcW = (const float*)d_in[11];
    const float* fcb = (const float*)d_in[12];

    const int n = in_sizes[0] / 5;
    const int E = in_sizes[1] / 2;
    const int* src = edge;
    const int* dst = edge + E;

    // Workspace layout (f32 units): bufA | bufB | xpad | dinv | pooled | counts | Wp
    float* ws = (float*)d_ws;
    const size_t maxNd = (size_t)n * D_FINAL;
    float*  bufA   = ws;
    float*  bufB   = bufA + maxNd;
    float*  xpad   = bufB + maxNd;                       // n x 32 f32
    float*  dinv   = xpad + (size_t)n * 32;
    float*  pooled = dinv + n;
    float*  counts = pooled + (size_t)N_GRAPHS * D_FINAL;
    size_t  tail   = (size_t)(counts + N_GRAPHS - ws);
    tail = (tail + 15) & ~(size_t)15;                    // 64B align for bf16 vec loads
    __bf16* Wp     = (__bf16*)(ws + tail);               // <= 32768 bf16 (64 KB)

    auto cdiv = [](long long a, long long b) -> unsigned { return (unsigned)((a + b - 1) / b); };

    // Degrees (self-loop gives deg >= 1), then dinv = rsqrt(deg) in place.
    k_fill<<<cdiv(n, 256), 256, 0, stream>>>(dinv, 1.0f, n);
    k_deg_acc<<<cdiv(E, 256), 256, 0, stream>>>(dst, dinv, E);
    k_rsqrt_inplace<<<cdiv(n, 256), 256, 0, stream>>>(dinv, n);

    // Layer-1 input: pad 5 -> 32 columns.
    k_pad_x<<<cdiv((long long)n * 32, 256), 256, 0, stream>>>(x, xpad, n);

    const int dims_out[4] = {32, 64, 128, 256};
    const float* hin = xpad;
    for (int L = 0; L < 4; ++L) {
        const int d = dims_out[L];
        const long long tiles = (long long)(n >> 4) * (d >> 4);
        const unsigned gblk = cdiv(tiles, WAVES_PER_BLOCK);
        switch (L) {  // pack W into B-operand layout, then WMMA GEMM -> bufA
        case 0:
            k_pack_w<32, 32><<<cdiv(1 * 2 * 512, 256), 256, 0, stream>>>(Wl[0], Wp, 5);
            k_gemm_wmma_bf16<32, 32><<<gblk, WAVES_PER_BLOCK * 32, 0, stream>>>(hin, Wp, bufA, n);
            break;
        case 1:
            k_pack_w<32, 64><<<cdiv(1 * 4 * 512, 256), 256, 0, stream>>>(Wl[1], Wp, 32);
            k_gemm_wmma_bf16<32, 64><<<gblk, WAVES_PER_BLOCK * 32, 0, stream>>>(hin, Wp, bufA, n);
            break;
        case 2:
            k_pack_w<64, 128><<<cdiv(2 * 8 * 512, 256), 256, 0, stream>>>(Wl[2], Wp, 64);
            k_gemm_wmma_bf16<64, 128><<<gblk, WAVES_PER_BLOCK * 32, 0, stream>>>(hin, Wp, bufA, n);
            break;
        default:
            k_pack_w<128, 256><<<cdiv(4 * 16 * 512, 256), 256, 0, stream>>>(Wl[3], Wp, 128);
            k_gemm_wmma_bf16<128, 256><<<gblk, WAVES_PER_BLOCK * 32, 0, stream>>>(hin, Wp, bufA, n);
            break;
        }
        const long long nd4 = (long long)n * (d >> 2);
        const long long ed4 = (long long)E * (d >> 2);
        k_self_init<<<cdiv(nd4, 256), 256, 0, stream>>>(bufA, dinv, bufB, n, d);
        k_scatter<<<cdiv(ed4, 256), 256, 0, stream>>>(src, dst, dinv, bufA, bufB, E, d);
        k_bias_relu<<<cdiv(nd4, 256), 256, 0, stream>>>(bufB, bl[L], n, d);
        hin = bufB;
    }

    // Global mean pool + classifier head.
    k_fill<<<cdiv(N_GRAPHS * D_FINAL + N_GRAPHS, 256), 256, 0, stream>>>(
        pooled, 0.0f, N_GRAPHS * D_FINAL + N_GRAPHS);
    k_count<<<cdiv(n, 256), 256, 0, stream>>>(batch, counts, n);
    k_pool<<<cdiv((long long)n * (D_FINAL >> 2), 256), 256, 0, stream>>>(batch, bufB, pooled, n);
    k_head<<<1, N_GRAPHS, 0, stream>>>(pooled, counts, fcW, fcb, (float*)d_out);
}
